// TransformerEncoder_27358941676316
// MI455X (gfx1250) — matrix operand order repaired and checked by reading, not compiled, not run on hardware
//
#include <hip/hip_runtime.h>
#include <math.h>

// ---------------- problem dims (compile-time) ----------------
#define Bd   16
#define Sd   512
#define Hd   768
#define NHd  12
#define FFd  3072
#define Ld   4
#define Id   64
#define DHd  64
#define BSd  (Bd * Sd)       // 8192 rows
#define BNH  (Bd * NHd)      // 192 attention batches

// ---------------- types ----------------
typedef __attribute__((ext_vector_type(16))) __bf16   v16bf;
typedef __attribute__((ext_vector_type(8)))  float    v8f;
typedef __attribute__((ext_vector_type(4)))  unsigned int u32x4;
typedef int v4i_vs __attribute__((vector_size(16)));   // matches builtin param type

// ---------------- CDNA5 async global->LDS path ----------------
#if defined(__HIP_DEVICE_COMPILE__)
#  if __has_builtin(__builtin_amdgcn_global_load_async_to_lds_b128)
#    define USE_ASYNC_LDS 1
#  else
#    define USE_ASYNC_LDS 0
#  endif
#else
#  define USE_ASYNC_LDS 0
#endif

union Frag {
    v16bf v;
    u32x4 x4[2];
};

__device__ __forceinline__ void async_copy16(const void* g, void* l) {
#if USE_ASYNC_LDS
    __builtin_amdgcn_global_load_async_to_lds_b128(
        (__attribute__((address_space(1))) v4i_vs*)(g),
        (__attribute__((address_space(3))) v4i_vs*)(l),
        0, 0);
#else
    *(u32x4*)l = *(const u32x4*)g;
#endif
}

__device__ __forceinline__ void wait_async_done() {
#if USE_ASYNC_LDS
#  if __has_builtin(__builtin_amdgcn_s_wait_asynccnt)
    __builtin_amdgcn_s_wait_asynccnt(0);
#  else
    asm volatile("s_wait_asynccnt 0x0" ::: "memory");
#  endif
#endif
}

// =====================================================================
// Tiled GEMM:  C[M,N] = A[M,K] @ BT[N,K]^T  (+bias) (+gelu)
// A, BT row-major bf16.  C fp32 (optional), Cbf bf16 (optional).
// grid = (N/BN, M/128, batch); block = 256 threads = 8 waves.
// Wave (wm 0..3, wn 0..1): rows [wm*32,+32) x cols [wn*(BN/2),+BN/2)
// as 2 x (BN/32) grid of 16x16 WMMA tiles, K-step = 32.
// Double-buffered LDS staged via GLOBAL_LOAD_ASYNC_TO_LDS_B128.
// All M % 128 == 0, N % BN == 0, K % 32 == 0 by construction.
// =====================================================================
#define LDA 40   // 32 bf16 + 8 pad (80B rows, 16B aligned)

__device__ __forceinline__ float gelu_tanh(float x) {
    float x3 = x * x * x;
    return 0.5f * x * (1.0f + tanhf(0.79788456080286535588f * (x + 0.044715f * x3)));
}

template<int BN>
__global__ __launch_bounds__(256)
void gemm_bf16_wmma(const __bf16* __restrict__ A,
                    const __bf16* __restrict__ BT,
                    const float*  __restrict__ bias,
                    float*        __restrict__ C,
                    __bf16*       __restrict__ Cbf,
                    int M, int N, int K,
                    long long strideA, long long strideBT, long long strideC,
                    int act)
{
    constexpr int NT = BN / 32;              // n-tiles per wave (2 or 4)
    __shared__ __align__(16) __bf16 ldsA[2][128 * LDA];
    __shared__ __align__(16) __bf16 ldsB[2][BN * LDA];

    const int tid   = threadIdx.x;
    const int lane  = tid & 31;
    const int lrow  = lane & 15;
    const int lhalf = lane >> 4;
    const int w     = tid >> 5;              // wave 0..7
    const int wm    = w >> 1;                // 0..3
    const int wn    = w & 1;                 // 0..1

    const int rowBase = blockIdx.y * 128;
    const int colBase = blockIdx.x * BN;
    const size_t zA = (size_t)blockIdx.z * (size_t)strideA;
    const size_t zB = (size_t)blockIdx.z * (size_t)strideBT;
    const size_t zC = (size_t)blockIdx.z * (size_t)strideC;

    v8f acc[2][NT] = {};

    // stage one 32-wide K-slice of A (128 rows) and BT (BN rows) into buffer
    auto stage = [&](int buf, int kk) {
        #pragma unroll
        for (int it = 0; it < 2; ++it) {
            int idx = tid + it * 256;               // 0..511
            int r = idx >> 2;                       // 0..127
            int c = (idx & 3) << 3;                 // 0,8,16,24
            async_copy16(A + zA + (size_t)(rowBase + r) * K + kk + c,
                         &ldsA[buf][r * LDA + c]);
        }
        #pragma unroll
        for (int it = 0; it < BN / 64; ++it) {
            int idx = tid + it * 256;
            int r = idx >> 2;                       // 0..BN-1
            int c = (idx & 3) << 3;
            async_copy16(BT + zB + (size_t)(colBase + r) * K + kk + c,
                         &ldsB[buf][r * LDA + c]);
        }
    };

    const int nk = K >> 5;
    stage(0, 0);

    for (int t = 0; t < nk; ++t) {
        const int cur = t & 1;
        wait_async_done();
        __syncthreads();                 // buffer `cur` ready for all waves
        if (t + 1 < nk) stage(cur ^ 1, (t + 1) << 5);

        const unsigned int* pa = (const unsigned int*)&ldsA[cur][0];
        const unsigned int* pb = (const unsigned int*)&ldsB[cur][0];

        Frag a[2], b[NT];
        #pragma unroll
        for (int i = 0; i < 2; ++i) {
            int ar = wm * 32 + i * 16 + lrow;
            a[i].x4[0] = *(const u32x4*)&pa[ar * (LDA / 2) + lhalf * 4];
            a[i].x4[1] = *(const u32x4*)&pa[ar * (LDA / 2) + 8 + lhalf * 4];
        }
        #pragma unroll
        for (int j = 0; j < NT; ++j) {
            int br = wn * (BN / 2) + j * 16 + lrow;
            b[j].x4[0] = *(const u32x4*)&pb[br * (LDA / 2) + lhalf * 4];
            b[j].x4[1] = *(const u32x4*)&pb[br * (LDA / 2) + 8 + lhalf * 4];
        }

        #pragma unroll
        for (int i = 0; i < 2; ++i)
            #pragma unroll
            for (int j = 0; j < NT; ++j)
                acc[i][j] = __builtin_amdgcn_wmma_f32_16x16x32_bf16(
                    false, a[i].v, false, b[j].v,
                    (short)0, acc[i][j], false, false);
        // next iteration's barrier protects buffer reuse
    }

    // ---- epilogue ----
    #pragma unroll
    for (int i = 0; i < 2; ++i) {
        #pragma unroll
        for (int j = 0; j < NT; ++j) {
            int row0 = rowBase + wm * 32 + i * 16 + lhalf * 8;
            int col  = colBase + wn * (BN / 2) + j * 16 + lrow;
            float bval = bias ? bias[col] : 0.0f;
            #pragma unroll
            for (int r = 0; r < 8; ++r) {
                float v = acc[i][j][r] + bval;
                if (act) v = gelu_tanh(v);
                size_t off = zC + (size_t)(row0 + r) * N + col;
                if (C)   C[off]   = v;
                if (Cbf) Cbf[off] = (__bf16)v;
            }
        }
    }
}

// =====================================================================
// Weight transpose + fp32 -> bf16:  out[n*K+k] = in[k*N+n]
// =====================================================================
__global__ void transpose_cast_kernel(const float* __restrict__ in,
                                      __bf16* __restrict__ out,
                                      int K, int N)
{
    long long idx = (long long)blockIdx.x * 256 + threadIdx.x;
    long long total = (long long)K * N;
    if (idx >= total) return;
    int n = (int)(idx / K);
    int k = (int)(idx % K);
    out[idx] = (__bf16)in[(size_t)k * N + n];
}

// =====================================================================
// Embedding gather + positional add + LayerNorm. One block per token row.
// =====================================================================
__global__ __launch_bounds__(256)
void embed_ln_kernel(const int* __restrict__ ids,
                     const float* __restrict__ emb,
                     const float* __restrict__ pos,
                     const float* __restrict__ g,
                     const float* __restrict__ beta,
                     float* __restrict__ hout,
                     __bf16* __restrict__ hbf)
{
    int row = blockIdx.x;            // 0..BSd-1
    int s = row % Sd;
    int tok = ids[row];
    __shared__ float xb[Hd];
    __shared__ float red[256];
    int tid = threadIdx.x;

    float lsum = 0.0f;
    for (int c = tid; c < Hd; c += 256) {
        float x = emb[(size_t)tok * Hd + c] + pos[(size_t)s * Hd + c];
        xb[c] = x;
        lsum += x;
    }
    red[tid] = lsum; __syncthreads();
    for (int st = 128; st > 0; st >>= 1) { if (tid < st) red[tid] += red[tid + st]; __syncthreads(); }
    float mean = red[0] / (float)Hd; __syncthreads();

    float lvar = 0.0f;
    for (int c = tid; c < Hd; c += 256) { float d = xb[c] - mean; lvar += d * d; }
    red[tid] = lvar; __syncthreads();
    for (int st = 128; st > 0; st >>= 1) { if (tid < st) red[tid] += red[tid + st]; __syncthreads(); }
    float rstd = rsqrtf(red[0] / (float)Hd + 1e-12f);

    for (int c = tid; c < Hd; c += 256) {
        float y = (xb[c] - mean) * rstd * g[c] + beta[c];
        size_t off = (size_t)row * Hd + c;
        hout[off] = y;
        hbf[off]  = (__bf16)y;
    }
}

// =====================================================================
// h = LN(hin + delta). hout may alias hin (block touches only own row).
// =====================================================================
__global__ __launch_bounds__(256)
void residual_ln_kernel(const float* __restrict__ hin,
                        const float* __restrict__ delta,
                        const float* __restrict__ g,
                        const float* __restrict__ beta,
                        float* __restrict__ hout,
                        __bf16* __restrict__ hbf)
{
    int row = blockIdx.x;
    __shared__ float xb[Hd];
    __shared__ float red[256];
    int tid = threadIdx.x;

    float lsum = 0.0f;
    for (int c = tid; c < Hd; c += 256) {
        size_t off = (size_t)row * Hd + c;
        float x = hin[off] + delta[off];
        xb[c] = x; lsum += x;
    }
    red[tid] = lsum; __syncthreads();
    for (int st = 128; st > 0; st >>= 1) { if (tid < st) red[tid] += red[tid + st]; __syncthreads(); }
    float mean = red[0] / (float)Hd; __syncthreads();

    float lvar = 0.0f;
    for (int c = tid; c < Hd; c += 256) { float d = xb[c] - mean; lvar += d * d; }
    red[tid] = lvar; __syncthreads();
    for (int st = 128; st > 0; st >>= 1) { if (tid < st) red[tid] += red[tid + st]; __syncthreads(); }
    float rstd = rsqrtf(red[0] / (float)Hd + 1e-12f);

    for (int c = tid; c < Hd; c += 256) {
        float y = (xb[c] - mean) * rstd * g[c] + beta[c];
        size_t off = (size_t)row * Hd + c;
        hout[off] = y;
        hbf[off]  = (__bf16)y;
    }
}

// =====================================================================
// Split qkv rows into WMMA-friendly per-head buffers.
// q_bf,k_bf: [B,NH,S,DH]; vt_bf: [B,NH,DH,S]. q pre-scaled by 1/sqrt(DH).
// =====================================================================
__global__ void qkv_split_kernel(const float* __restrict__ qkv,
                                 __bf16* __restrict__ q,
                                 __bf16* __restrict__ k,
                                 __bf16* __restrict__ vt)
{
    int idx = blockIdx.x * 256 + threadIdx.x;
    if (idx >= Bd * NHd * Sd * DHd) return;
    int d = idx % DHd;
    int s = (idx / DHd) % Sd;
    int h = (idx / (DHd * Sd)) % NHd;
    int b = idx / (DHd * Sd * NHd);
    size_t rowoff = (size_t)(b * Sd + s) * (3 * Hd);
    int coff = h * DHd + d;
    q[idx] = (__bf16)(qkv[rowoff + coff] * 0.125f);
    k[idx] = (__bf16)(qkv[rowoff + Hd + coff]);
    vt[((size_t)(b * NHd + h) * DHd + d) * Sd + s] = (__bf16)(qkv[rowoff + 2 * Hd + coff]);
}

// =====================================================================
// Row softmax with additive mask bias. One block per (b,h,q) row of S=512.
// Reads fp32 scores, writes bf16 probabilities.
// =====================================================================
__global__ __launch_bounds__(256)
void softmax_kernel(const float* __restrict__ scores,
                    const int*   __restrict__ amask,
                    __bf16*      __restrict__ attn)
{
    int row = blockIdx.x;                    // b*NH*S + h*S + q
    int b = row / (NHd * Sd);
    const float* src = scores + (size_t)row * Sd;
    __bf16* dst = attn + (size_t)row * Sd;
    __shared__ float red[256];
    int tid = threadIdx.x;

    float bias0 = (1.0f - (float)amask[b * Sd + tid])       * -1e9f;
    float bias1 = (1.0f - (float)amask[b * Sd + tid + 256]) * -1e9f;
    float v0 = src[tid] + bias0;
    float v1 = src[tid + 256] + bias1;

    red[tid] = fmaxf(v0, v1); __syncthreads();
    for (int st = 128; st > 0; st >>= 1) { if (tid < st) red[tid] = fmaxf(red[tid], red[tid + st]); __syncthreads(); }
    float rowmax = red[0]; __syncthreads();

    float e0 = __expf(v0 - rowmax);
    float e1 = __expf(v1 - rowmax);
    red[tid] = e0 + e1; __syncthreads();
    for (int st = 128; st > 0; st >>= 1) { if (tid < st) red[tid] += red[tid + st]; __syncthreads(); }
    float inv = 1.0f / red[0];

    dst[tid]       = (__bf16)(e0 * inv);
    dst[tid + 256] = (__bf16)(e1 * inv);
}

// =====================================================================
// ctx [B,NH,S,DH] fp32 -> ctx_bf [B,S,H] bf16
// =====================================================================
__global__ void ctx_permute_kernel(const float* __restrict__ ctx,
                                   __bf16* __restrict__ out)
{
    int idx = blockIdx.x * 256 + threadIdx.x;
    if (idx >= Bd * NHd * Sd * DHd) return;
    int d = idx % DHd;
    int s = (idx / DHd) % Sd;
    int h = (idx / (DHd * Sd)) % NHd;
    int b = idx / (DHd * Sd * NHd);
    out[(size_t)(b * Sd + s) * Hd + h * DHd + d] = (__bf16)ctx[idx];
}

// =====================================================================
// Ragged span mean-pooling. One block per (b, instance).
// =====================================================================
__global__ __launch_bounds__(256)
void span_pool_kernel(const float* __restrict__ h,
                      const int* __restrict__ starts,
                      const int* __restrict__ ends,
                      const int* __restrict__ ninst,
                      float* __restrict__ out)
{
    int bi = blockIdx.x;                 // 0..B*I-1
    int b = bi / Id;
    int i = bi % Id;
    int st = starts[bi];
    int en = ends[bi];
    bool valid = (i < ninst[b]);
    int cnt = en - st; if (cnt < 1) cnt = 1;
    float inv = valid ? (1.0f / (float)cnt) : 0.0f;
    for (int c = threadIdx.x; c < Hd; c += 256) {
        float acc = 0.0f;
        if (valid)
            for (int s = st; s < en; ++s)
                acc += h[(size_t)(b * Sd + s) * Hd + c];
        out[(size_t)bi * Hd + c] = acc * inv;
    }
}

// =====================================================================
// Host driver
// =====================================================================
extern "C" void kernel_launch(void* const* d_in, const int* in_sizes, int n_in,
                              void* d_out, int out_size, void* d_ws, size_t ws_size,
                              hipStream_t stream)
{
    (void)in_sizes; (void)n_in; (void)out_size; (void)ws_size;

    const int*   ids    = (const int*)d_in[0];
    const int*   amask  = (const int*)d_in[1];
    const int*   starts = (const int*)d_in[2];
    const int*   ends   = (const int*)d_in[3];
    const int*   ninst  = (const int*)d_in[4];
    const float* emb    = (const float*)d_in[5];
    const float* pos    = (const float*)d_in[6];
    const float* eg     = (const float*)d_in[7];
    const float* eb     = (const float*)d_in[8];
    const float* Wqkv   = (const float*)d_in[9];
    const float* bqkv   = (const float*)d_in[10];
    const float* Wo     = (const float*)d_in[11];
    const float* bo     = (const float*)d_in[12];
    const float* ln1g   = (const float*)d_in[13];
    const float* ln1b   = (const float*)d_in[14];
    const float* W1     = (const float*)d_in[15];
    const float* b1     = (const float*)d_in[16];
    const float* W2     = (const float*)d_in[17];
    const float* b2     = (const float*)d_in[18];
    const float* ln2g   = (const float*)d_in[19];
    const float* ln2b   = (const float*)d_in[20];
    float* out = (float*)d_out;

    // ---- carve workspace ----
    char* ws = (char*)d_ws;
    size_t off = 0;
    auto carve = [&](size_t bytes) -> char* {
        char* p = ws + off;
        off += (bytes + 255) & ~(size_t)255;
        return p;
    };
    __bf16* wqkvT = (__bf16*)carve((size_t)Ld * 3 * Hd * Hd * 2);
    __bf16* woT   = (__bf16*)carve((size_t)Ld * Hd * Hd * 2);
    __bf16* w1T   = (__bf16*)carve((size_t)Ld * FFd * Hd * 2);
    __bf16* w2T   = (__bf16*)carve((size_t)Ld * Hd * FFd * 2);
    float*  h_f32 = (float*) carve((size_t)BSd * Hd * 4);
    __bf16* h_bf  = (__bf16*)carve((size_t)BSd * Hd * 2);
    float*  buf1  = (float*) carve((size_t)BSd * FFd * 4);   // qkv out / ff1 out
    __bf16* ff_bf = (__bf16*)carve((size_t)BSd * FFd * 2);
    float*  delta = (float*) carve((size_t)BSd * Hd * 4);    // wo out / ff2 out
    __bf16* q_bf  = (__bf16*)carve((size_t)BNH * Sd * DHd * 2);
    __bf16* k_bf  = (__bf16*)carve((size_t)BNH * Sd * DHd * 2);
    __bf16* vt_bf = (__bf16*)carve((size_t)BNH * Sd * DHd * 2);
    float*  scrs  = (float*) carve((size_t)BNH * Sd * Sd * 4);
    __bf16* attn  = (__bf16*)carve((size_t)BNH * Sd * Sd * 2);
    float*  ctx   = (float*) carve((size_t)BNH * Sd * DHd * 4);
    __bf16* ctxbf = (__bf16*)carve((size_t)BSd * Hd * 2);

    auto tgrid = [](long long total) { return dim3((unsigned)((total + 255) / 256)); };

    // ---- weights -> bf16 transposed ----
    for (int l = 0; l < Ld; ++l) {
        transpose_cast_kernel<<<tgrid((long long)Hd * 3 * Hd), 256, 0, stream>>>(
            Wqkv + (size_t)l * Hd * 3 * Hd, wqkvT + (size_t)l * 3 * Hd * Hd, Hd, 3 * Hd);
        transpose_cast_kernel<<<tgrid((long long)Hd * Hd), 256, 0, stream>>>(
            Wo + (size_t)l * Hd * Hd, woT + (size_t)l * Hd * Hd, Hd, Hd);
        transpose_cast_kernel<<<tgrid((long long)Hd * FFd), 256, 0, stream>>>(
            W1 + (size_t)l * Hd * FFd, w1T + (size_t)l * FFd * Hd, Hd, FFd);
        transpose_cast_kernel<<<tgrid((long long)FFd * Hd), 256, 0, stream>>>(
            W2 + (size_t)l * FFd * Hd, w2T + (size_t)l * Hd * FFd, FFd, Hd);
    }

    // ---- embeddings + LN ----
    embed_ln_kernel<<<BSd, 256, 0, stream>>>(ids, emb, pos, eg, eb, h_f32, h_bf);

    const long long sQK = (long long)Sd * DHd;
    const long long sSS = (long long)Sd * Sd;

    for (int l = 0; l < Ld; ++l) {
        // qkv = h @ Wqkv + bqkv                    [8192 x 2304, K=768]
        gemm_bf16_wmma<128><<<dim3(3 * Hd / 128, BSd / 128, 1), 256, 0, stream>>>(
            h_bf, wqkvT + (size_t)l * 3 * Hd * Hd, bqkv + (size_t)l * 3 * Hd,
            buf1, nullptr, BSd, 3 * Hd, Hd, 0, 0, 0, 0);

        qkv_split_kernel<<<tgrid((long long)Bd * NHd * Sd * DHd), 256, 0, stream>>>(
            buf1, q_bf, k_bf, vt_bf);

        // scores = (q/8) @ k^T   batched over B*NH  [512 x 512, K=64]
        gemm_bf16_wmma<128><<<dim3(Sd / 128, Sd / 128, BNH), 256, 0, stream>>>(
            q_bf, k_bf, nullptr, scrs, nullptr, Sd, Sd, DHd, sQK, sQK, sSS, 0);

        softmax_kernel<<<BNH * Sd, 256, 0, stream>>>(scrs, amask, attn);

        // ctx = attn @ v   batched  [512 x 64, K=512]  (narrow N -> BN=64)
        gemm_bf16_wmma<64><<<dim3(DHd / 64, Sd / 128, BNH), 256, 0, stream>>>(
            attn, vt_bf, nullptr, ctx, nullptr, Sd, DHd, Sd, sSS, sQK, sQK, 0);

        ctx_permute_kernel<<<tgrid((long long)Bd * NHd * Sd * DHd), 256, 0, stream>>>(ctx, ctxbf);

        // attn_out = ctx @ Wo + bo                 [8192 x 768, K=768]
        gemm_bf16_wmma<128><<<dim3(Hd / 128, BSd / 128, 1), 256, 0, stream>>>(
            ctxbf, woT + (size_t)l * Hd * Hd, bo + (size_t)l * Hd,
            delta, nullptr, BSd, Hd, Hd, 0, 0, 0, 0);

        residual_ln_kernel<<<BSd, 256, 0, stream>>>(
            h_f32, delta, ln1g + (size_t)l * Hd, ln1b + (size_t)l * Hd, h_f32, h_bf);

        // ff1 = gelu(h @ W1 + b1)                  [8192 x 3072, K=768], bf16 side-output
        gemm_bf16_wmma<128><<<dim3(FFd / 128, BSd / 128, 1), 256, 0, stream>>>(
            h_bf, w1T + (size_t)l * FFd * Hd, b1 + (size_t)l * FFd,
            buf1, ff_bf, BSd, FFd, Hd, 0, 0, 0, 1);

        // ff2 = ff1 @ W2 + b2                      [8192 x 768, K=3072]
        gemm_bf16_wmma<128><<<dim3(Hd / 128, BSd / 128, 1), 256, 0, stream>>>(
            ff_bf, w2T + (size_t)l * Hd * FFd, b2 + (size_t)l * Hd,
            delta, nullptr, BSd, Hd, FFd, 0, 0, 0, 0);

        residual_ln_kernel<<<BSd, 256, 0, stream>>>(
            h_f32, delta, ln2g + (size_t)l * Hd, ln2b + (size_t)l * Hd, h_f32, h_bf);
    }

    // ---- span mean pooling ----
    span_pool_kernel<<<Bd * Id, 256, 0, stream>>>(h_f32, starts, ends, ninst, out);
}
